// TemporalPropagator_62646392979648
// MI455X (gfx1250) — compile-verified, matrix operand order checked by reading, not run on hardware
//
#include <hip/hip_runtime.h>

// ---------------------------------------------------------------------------
// TemporalPropagator for MI455X (gfx1250): WMMA f16 GEMM pipeline,
// double-buffered LDS with async global->LDS B-tile staging.
//   D=512, B=8, T=2048  ->  M = 16384 rows
// ---------------------------------------------------------------------------

#define D 512
#define BB 8
#define TT 2048
#define MM (BB * TT)          // 16384
#define D2 (2 * D)            // 1024

typedef __attribute__((ext_vector_type(16))) _Float16 v16h;
typedef __attribute__((ext_vector_type(8)))  float    v8f;
typedef __attribute__((ext_vector_type(4)))  int      v4i;

struct alignas(16) h8 { _Float16 v[8]; };

#if defined(__gfx1250__) && \
    __has_builtin(__builtin_amdgcn_global_load_async_to_lds_b128) && \
    __has_builtin(__builtin_amdgcn_s_wait_asynccnt)
#define USE_ASYNC_LDS 1
#define AS1 __attribute__((address_space(1)))
#define AS3 __attribute__((address_space(3)))
#else
#define USE_ASYNC_LDS 0
#endif

__device__ __forceinline__ void cdiv(float ar, float ai, float br, float bi,
                                     float& cr, float& ci) {
  float inv = 1.0f / (br * br + bi * bi);
  cr = (ar * br + ai * bi) * inv;
  ci = (ai * br - ar * bi) * inv;
}

// ------------------------- small setup kernels -----------------------------

// M = I + (A - A^H), R = I - (A - A^H), A = vre + i*vim
__global__ void build_skew_kernel(const float* __restrict__ vre,
                                  const float* __restrict__ vim,
                                  float* Mre, float* Mim, float* Rre, float* Rim) {
  int idx = blockIdx.x * blockDim.x + threadIdx.x;
  if (idx >= D * D) return;
  int i = idx >> 9, j = idx & (D - 1);
  float sre = vre[i * D + j] - vre[j * D + i];
  float sim = vim[i * D + j] + vim[j * D + i];
  float id = (i == j) ? 1.0f : 0.0f;
  Mre[idx] = id + sre;  Mim[idx] = sim;
  Rre[idx] = id - sre;  Rim[idx] = -sim;
}

// factors f[i] = M[i][k] / M[k][k]  (f[k] forced to 0)
__global__ void gj_factor_kernel(const float* __restrict__ Mre,
                                 const float* __restrict__ Mim,
                                 float* fre, float* fim, int k) {
  int i = threadIdx.x;
  float pr = Mre[k * D + k], pi = Mim[k * D + k];
  float fr, fi;
  cdiv(Mre[i * D + k], Mim[i * D + k], pr, pi, fr, fi);
  if (i == k) { fr = 0.0f; fi = 0.0f; }
  fre[i] = fr; fim[i] = fi;
}

// rows i != k:  row_i -= f[i] * row_k   (both M and R)
__global__ void gj_elim_kernel(float* Mre, float* Mim, float* Rre, float* Rim,
                               const float* __restrict__ fre,
                               const float* __restrict__ fim, int k) {
  int idx = blockIdx.x * blockDim.x + threadIdx.x;
  if (idx >= D * D) return;
  int i = idx >> 9, j = idx & (D - 1);
  if (i == k) return;
  float fr = fre[i], fi = fim[i];
  float mr = Mre[k * D + j], mi = Mim[k * D + j];
  Mre[idx] -= fr * mr - fi * mi;
  Mim[idx] -= fr * mi + fi * mr;
  float rr = Rre[k * D + j], ri = Rim[k * D + j];
  Rre[idx] -= fr * rr - fi * ri;
  Rim[idx] -= fr * ri + fi * rr;
}

// W = (R / diag(M)) * S[col] * (1-alpha) + DFT * alpha
__global__ void finish_W_kernel(const float* __restrict__ Mre,
                                const float* __restrict__ Mim,
                                const float* __restrict__ Rre,
                                const float* __restrict__ Rim,
                                const float* __restrict__ log_sigma,
                                const float* __restrict__ dftw,
                                float* Wre, float* Wim) {
  int idx = blockIdx.x * blockDim.x + threadIdx.x;
  if (idx >= D * D) return;
  int i = idx >> 9, j = idx & (D - 1);
  float vr, vi;
  cdiv(Rre[idx], Rim[idx], Mre[i * D + i], Mim[i * D + i], vr, vi);
  float s = __expf(log_sigma[j]);
  float alpha = 1.0f / (1.0f + __expf(-dftw[0]));
  int p = (i * j) & (D - 1);
  float ang = -6.2831853071795864769f * (float)p / (float)D;
  const float rsq = 0.044194173824159220275f; // 1/sqrt(512)
  float dr = __cosf(ang) * rsq, di = __sinf(ang) * rsq;
  Wre[idx] = vr * s * (1.0f - alpha) + dr * alpha;
  Wim[idx] = vi * s * (1.0f - alpha) + di * alpha;
}

__global__ void init_inv_kernel(const float* __restrict__ Wre,
                                const float* __restrict__ Wim,
                                float* Mre, float* Mim, float* Rre, float* Rim) {
  int idx = blockIdx.x * blockDim.x + threadIdx.x;
  if (idx >= D * D) return;
  int i = idx >> 9, j = idx & (D - 1);
  Mre[idx] = Wre[idx]; Mim[idx] = Wim[idx];
  Rre[idx] = (i == j) ? 1.0f : 0.0f; Rim[idx] = 0.0f;
}

__global__ void finish_inv_kernel(const float* __restrict__ Mre,
                                  const float* __restrict__ Mim,
                                  const float* __restrict__ Rre,
                                  const float* __restrict__ Rim,
                                  float* Ire, float* Iim) {
  int idx = blockIdx.x * blockDim.x + threadIdx.x;
  if (idx >= D * D) return;
  int i = idx >> 9;
  float vr, vi;
  cdiv(Rre[idx], Rim[idx], Mre[i * D + i], Mim[i * D + i], vr, vi);
  Ire[idx] = vr; Iim[idx] = vi;
}

// Wt[n][k] = W[k][n]  (f16, B-operand layout for the WMMA GEMM)
__global__ void transposeW_f16_kernel(const float* __restrict__ Wre,
                                      const float* __restrict__ Wim,
                                      _Float16* Tre, _Float16* Tim) {
  int idx = blockIdx.x * blockDim.x + threadIdx.x;
  if (idx >= D * D) return;
  int n = idx >> 9, k = idx & (D - 1);
  Tre[n * D + k] = (_Float16)Wre[k * D + n];
  Tim[n * D + k] = (_Float16)Wim[k * D + n];
}

// Winvcat[n][k] = k<512 ? Winv_re[k][n] : -Winv_im[k-512][n]   (512 x 1024)
__global__ void winvcat_f16_kernel(const float* __restrict__ Ire,
                                   const float* __restrict__ Iim,
                                   _Float16* Cat) {
  int idx = blockIdx.x * blockDim.x + threadIdx.x;
  if (idx >= D * D2) return;
  int n = idx >> 10, k = idx & (D2 - 1);
  float v = (k < D) ? Ire[k * D + n] : -Iim[(k - D) * D + n];
  Cat[n * D2 + k] = (_Float16)v;
}

__global__ void cvt_f16_kernel(const float* __restrict__ in, _Float16* out, int n) {
  int idx = blockIdx.x * blockDim.x + threadIdx.x;
  if (idx < n) out[idx] = (_Float16)in[idx];
}

// per-channel t_decay / forcing: tdf[4d] = {t_re, t_im, f_re, f_im}
__global__ void tdf_kernel(const float* __restrict__ lam_re,
                           const float* __restrict__ lam_im, float* tdf) {
  int d = blockIdx.x * blockDim.x + threadIdx.x;
  if (d >= D) return;
  float lr = fminf(fmaxf(lam_re[d], -0.3f), 0.3f);
  float li = lam_im[d];
  float e = __expf(lr);
  float tr = e * __cosf(li), ti = e * __sinf(li);
  float sv = lr + 1e-12f;
  float sgn = (sv > 0.0f) ? 1.0f : ((sv < 0.0f) ? -1.0f : 0.0f);
  float lsr = lr + 1e-8f * sgn, lsi = li;
  float fr, fi;
  cdiv(tr - 1.0f, ti, lsr, lsi, fr, fi);
  tdf[4 * d] = tr; tdf[4 * d + 1] = ti; tdf[4 * d + 2] = fr; tdf[4 * d + 3] = fi;
}

// ------------------------- associative scan (3 pass) -----------------------
// states[t] = decay * states[t-1] + X[t], decay constant per channel.
#define CHUNK 64
#define NCH   (TT / CHUNK)   // 32

__global__ void scan_pass1(const float* __restrict__ Xre,
                           const float* __restrict__ Xim,
                           const float* __restrict__ dre,
                           const float* __restrict__ dim_, float2* carry) {
  int idx = blockIdx.x * blockDim.x + threadIdx.x;     // BB*D*NCH = 131072
  if (idx >= BB * D * NCH) return;
  int chunk = idx & (NCH - 1);
  int ch = idx >> 5;
  int d = ch & (D - 1), b = ch >> 9;
  float ar = 1.0f / (1.0f + __expf(-dre[d])), ai = dim_[d];
  size_t base = ((size_t)b * TT + (size_t)chunk * CHUNK) * D + d;
  float sr = 0.0f, si = 0.0f;
  for (int j = 0; j < CHUNK; ++j) {
    float xr = Xre[base + (size_t)j * D], xi = Xim[base + (size_t)j * D];
    float nr = sr * ar - si * ai + xr;
    float ni = sr * ai + si * ar + xi;
    sr = nr; si = ni;
  }
  carry[idx] = make_float2(sr, si);
}

__global__ void scan_pass2(const float* __restrict__ dre,
                           const float* __restrict__ dim_, float2* carry) {
  int ch = blockIdx.x * blockDim.x + threadIdx.x;      // BB*D = 4096
  if (ch >= BB * D) return;
  int d = ch & (D - 1);
  float pr = 1.0f / (1.0f + __expf(-dre[d])), pi = dim_[d];
  // decay^CHUNK by repeated squaring (CHUNK=64 = 2^6)
  for (int t = 0; t < 6; ++t) {
    float nr = pr * pr - pi * pi, ni = 2.0f * pr * pi;
    pr = nr; pi = ni;
  }
  float rr = 0.0f, ri = 0.0f;
  for (int c = 0; c < NCH; ++c) {
    float2 t = carry[ch * NCH + c];
    carry[ch * NCH + c] = make_float2(rr, ri);   // carry-in for chunk c
    float nr = rr * pr - ri * pi + t.x;
    float ni = rr * pi + ri * pr + t.y;
    rr = nr; ri = ni;
  }
}

__global__ void scan_pass3(float* Xre, float* Xim,
                           const float* __restrict__ dre,
                           const float* __restrict__ dim_,
                           const float2* __restrict__ carry) {
  int idx = blockIdx.x * blockDim.x + threadIdx.x;
  if (idx >= BB * D * NCH) return;
  int chunk = idx & (NCH - 1);
  int ch = idx >> 5;
  int d = ch & (D - 1), b = ch >> 9;
  float ar = 1.0f / (1.0f + __expf(-dre[d])), ai = dim_[d];
  float2 c0 = carry[idx];
  float sr = c0.x, si = c0.y;
  size_t base = ((size_t)b * TT + (size_t)chunk * CHUNK) * D + d;
  for (int j = 0; j < CHUNK; ++j) {
    size_t o = base + (size_t)j * D;
    float xr = Xre[o], xi = Xim[o];
    float nr = sr * ar - si * ai + xr;
    float ni = sr * ai + si * ar + xi;
    sr = nr; si = ni;
    Xre[o] = sr; Xim[o] = si;     // states in place
  }
}

// ------------------------------- WMMA GEMM ---------------------------------
// C[p] = A @ B[p] (+bias) with A = [A0|A1] concat over K (f32 -> f16),
// B[p] f16 stored [n][k] (row = output col, k contiguous) -> identical
// fragment addressing for A and B per the CDNA5 16-bit layouts.
// Block tile 128(M) x 64(N), K step 32, 8 waves, each wave 32x32 per plane.
// Double-buffered LDS; B tiles staged with async global->LDS (ASYNCcnt).

__device__ __forceinline__ v16h load_frag(const _Float16* tile, int lane) {
  const int row = lane & 15;
  const int k0 = (lane & 16) ? 8 : 0;
  const _Float16* p = tile + row * 32 + k0;
  v16h r;
#pragma unroll
  for (int i = 0; i < 8; ++i) r[i] = p[i];
#pragma unroll
  for (int i = 0; i < 8; ++i) r[8 + i] = p[16 + i];
  return r;
}

template <int APLANES, int OPLANES, int EPI>
__global__ __launch_bounds__(256) void gemm_wmma(
    const float* __restrict__ A0, const float* __restrict__ A1,
    const _Float16* __restrict__ B0, const _Float16* __restrict__ B1,
    const float* __restrict__ bias0, const float* __restrict__ bias1,
    float* __restrict__ C0, float* __restrict__ C1,
    int K, int N,
    const float* __restrict__ hre, const float* __restrict__ him,
    const float* __restrict__ tdf) {
  constexpr int BM = 128, BN = 64, BK = 32;
  __shared__ _Float16 As[2][BM * BK];
  __shared__ _Float16 Bs[2][OPLANES][BN * BK];

  const int tid = threadIdx.x;
  const int lane = tid & 31;
  const int wave = tid >> 5;
  const int wm = wave >> 1, wn = wave & 1;
  const int mBlk = blockIdx.y * BM;
  const int nBlk = blockIdx.x * BN;
  const int K0 = (APLANES == 2) ? (K >> 1) : K;

  v8f acc[OPLANES][2][2] = {};

  // per-thread staging registers
  float4 av[4];
#if !USE_ASYNC_LDS
  h8 bv[2];
#endif
  const int br = tid >> 2, bc8 = (tid & 3) * 8;   // B tile: one b128 per thread

  // issue all global loads (A -> regs, B -> LDS async) for tile kt
  auto stage_loads = [&](int kt, int buf) {
    const float* Asrc = A0;
    int kq = kt;
    if (APLANES == 2 && kt >= K0) { Asrc = A1; kq = kt - K0; }
#pragma unroll
    for (int i = 0; i < 4; ++i) {
      int idx = tid + i * 256;              // 1024 float4 slots (128 x 8)
      int r = idx >> 3, c4 = idx & 7;
      av[i] = *(const float4*)(Asrc + (size_t)(mBlk + r) * K0 + kq + c4 * 4);
    }
#if USE_ASYNC_LDS
    __builtin_amdgcn_global_load_async_to_lds_b128(
        (AS1 v4i*)(_Float16*)(B0 + (size_t)(nBlk + br) * K + kt + bc8),
        (AS3 v4i*)&Bs[buf][0][br * BK + bc8], 0, 0);
    if constexpr (OPLANES == 2) {
      __builtin_amdgcn_global_load_async_to_lds_b128(
          (AS1 v4i*)(_Float16*)(B1 + (size_t)(nBlk + br) * K + kt + bc8),
          (AS3 v4i*)&Bs[buf][1][br * BK + bc8], 0, 0);
    }
#else
    bv[0] = *(const h8*)(B0 + (size_t)(nBlk + br) * K + kt + bc8);
    if constexpr (OPLANES == 2)
      bv[1] = *(const h8*)(B1 + (size_t)(nBlk + br) * K + kt + bc8);
#endif
    if (kt + BK < K)
      __builtin_prefetch(Asrc + (size_t)(mBlk + (tid >> 1)) * K0 + kq + BK, 0, 3);
  };

  // convert + store staged A tile (and B in sync fallback) into LDS buffer
  auto stage_store = [&](int buf) {
#pragma unroll
    for (int i = 0; i < 4; ++i) {
      int idx = tid + i * 256;
      int r = idx >> 3, c4 = idx & 7;
      _Float16* dst = &As[buf][r * BK + c4 * 4];
      dst[0] = (_Float16)av[i].x; dst[1] = (_Float16)av[i].y;
      dst[2] = (_Float16)av[i].z; dst[3] = (_Float16)av[i].w;
    }
#if !USE_ASYNC_LDS
    *(h8*)&Bs[buf][0][br * BK + bc8] = bv[0];
    if constexpr (OPLANES == 2)
      *(h8*)&Bs[buf][1][br * BK + bc8] = bv[1];
#endif
  };

  // ---------------- pipelined main loop ----------------
  stage_loads(0, 0);
  stage_store(0);
  const int NIT = K / BK;
  for (int it = 0; it < NIT; ++it) {
    const int cur = it & 1;
#if USE_ASYNC_LDS
    __builtin_amdgcn_s_wait_asynccnt(0);   // our B tile for `cur` has landed
#endif
    __syncthreads();                        // all staging for `cur` visible
    const bool more = (it + 1 < NIT);
    if (more) stage_loads((it + 1) * BK, cur ^ 1);  // overlap with compute

    v16h afr0 = load_frag(&As[cur][(wm * 32 + 0) * BK], lane);
    v16h afr1 = load_frag(&As[cur][(wm * 32 + 16) * BK], lane);
#pragma unroll
    for (int p = 0; p < OPLANES; ++p) {
      v16h bfr0 = load_frag(&Bs[cur][p][(wn * 32 + 0) * BK], lane);
      v16h bfr1 = load_frag(&Bs[cur][p][(wn * 32 + 16) * BK], lane);
      acc[p][0][0] = __builtin_amdgcn_wmma_f32_16x16x32_f16(
          false, afr0, false, bfr0, (short)0, acc[p][0][0], false, false);
      acc[p][0][1] = __builtin_amdgcn_wmma_f32_16x16x32_f16(
          false, afr0, false, bfr1, (short)0, acc[p][0][1], false, false);
      acc[p][1][0] = __builtin_amdgcn_wmma_f32_16x16x32_f16(
          false, afr1, false, bfr0, (short)0, acc[p][1][0], false, false);
      acc[p][1][1] = __builtin_amdgcn_wmma_f32_16x16x32_f16(
          false, afr1, false, bfr1, (short)0, acc[p][1][1], false, false);
    }
    if (more) stage_store(cur ^ 1);   // cvt+store hides under WMMA issue
  }

  // ---- epilogue / store (C layout: lanes 0-15 rows 0-7, lanes 16-31 rows 8-15)
  const int col = lane & 15;
  const int rbase = (lane & 16) ? 8 : 0;
#pragma unroll
  for (int i = 0; i < 2; ++i) {
#pragma unroll
    for (int j = 0; j < 2; ++j) {
      int n = nBlk + wn * 32 + j * 16 + col;
      float b0 = bias0 ? bias0[n] : 0.0f;
      float b1 = 0.0f;
      if constexpr (OPLANES == 2) b1 = bias1 ? bias1[n] : 0.0f;
      float tr = 0, ti = 0, fr = 0, fi = 0;
      if constexpr (EPI == 1) {
        tr = tdf[4 * n]; ti = tdf[4 * n + 1];
        fr = tdf[4 * n + 2]; fi = tdf[4 * n + 3];
      }
#pragma unroll
      for (int r = 0; r < 8; ++r) {
        int m = mBlk + wm * 32 + i * 16 + rbase + r;
        size_t o = (size_t)m * N + n;
        float c0 = acc[0][i][j][r] + b0;
        if constexpr (OPLANES == 2) {
          float c1 = acc[1][i][j][r] + b1;
          if constexpr (EPI == 1) {
            float hr = hre[o], hi = him[o];
            C0[o] = hr * tr - hi * ti + c0 * fr - c1 * fi;
            C1[o] = hr * ti + hi * tr + c0 * fi + c1 * fr;
          } else {
            C0[o] = c0; C1[o] = c1;
          }
        } else {
          C0[o] = c0;
        }
      }
    }
  }
}

// ------------------------------ launch -------------------------------------

extern "C" void kernel_launch(void* const* d_in, const int* in_sizes, int n_in,
                              void* d_out, int out_size, void* d_ws, size_t ws_size,
                              hipStream_t stream) {
  const float* x        = (const float*)d_in[0];
  const float* vre      = (const float*)d_in[1];
  const float* vim      = (const float*)d_in[2];
  const float* log_sig  = (const float*)d_in[3];
  const float* dftw     = (const float*)d_in[4];
  const float* dec_re   = (const float*)d_in[5];
  const float* dec_im   = (const float*)d_in[6];
  const float* mix_w    = (const float*)d_in[7];
  const float* mix_b    = (const float*)d_in[8];
  const float* proj_w   = (const float*)d_in[9];
  const float* proj_b   = (const float*)d_in[10];
  const float* lam_re   = (const float*)d_in[11];
  const float* lam_im   = (const float*)d_in[12];
  float* out = (float*)d_out;

  char* wsp = (char*)d_ws;
  auto alloc = [&](size_t bytes) -> char* {
    char* p = wsp; wsp += (bytes + 255) & ~(size_t)255; return p;
  };
  const size_t DD = (size_t)D * D * sizeof(float);          // 1 MB
  const size_t MD = (size_t)MM * D * sizeof(float);         // 32 MB

  float* Wre  = (float*)alloc(DD);
  float* Wim  = (float*)alloc(DD);
  float* Ire  = (float*)alloc(DD);   // Winv
  float* Iim  = (float*)alloc(DD);
  float* Mre  = (float*)alloc(DD);   // Gauss-Jordan scratch
  float* Mim  = (float*)alloc(DD);
  float* Rre  = (float*)alloc(DD);
  float* Rim  = (float*)alloc(DD);
  float* fre  = (float*)alloc(D * sizeof(float));
  float* fim  = (float*)alloc(D * sizeof(float));
  _Float16* WTre   = (_Float16*)alloc((size_t)D * D * 2);
  _Float16* WTim   = (_Float16*)alloc((size_t)D * D * 2);
  _Float16* mixh   = (_Float16*)alloc((size_t)D2 * D2 * 2);
  _Float16* projh  = (_Float16*)alloc((size_t)D2 * D2 * 2);
  _Float16* icath  = (_Float16*)alloc((size_t)D * D2 * 2);
  float*  tdf   = (float*)alloc(4 * D * sizeof(float));
  float2* carry = (float2*)alloc((size_t)BB * D * NCH * sizeof(float2));
  float* h_re   = (float*)alloc(MD);
  float* h_im   = (float*)alloc(MD);
  float* X_re   = (float*)alloc(MD);   // mix output -> states (in place)
  float* X_im   = (float*)alloc(MD);
  float* ho_re  = (float*)alloc(MD);
  float* ho_im  = (float*)alloc(MD);

  const dim3 blk(256);
  const dim3 gDD((D * D + 255) / 256);

  // ---- 1) Cayley solve: V = (I+S)^-1 (I-S), then W blend ----
  build_skew_kernel<<<gDD, blk, 0, stream>>>(vre, vim, Mre, Mim, Rre, Rim);
  for (int k = 0; k < D; ++k) {
    gj_factor_kernel<<<1, D, 0, stream>>>(Mre, Mim, fre, fim, k);
    gj_elim_kernel<<<gDD, blk, 0, stream>>>(Mre, Mim, Rre, Rim, fre, fim, k);
  }
  finish_W_kernel<<<gDD, blk, 0, stream>>>(Mre, Mim, Rre, Rim, log_sig, dftw,
                                           Wre, Wim);

  // ---- 2) W_inv via Gauss-Jordan ----
  init_inv_kernel<<<gDD, blk, 0, stream>>>(Wre, Wim, Mre, Mim, Rre, Rim);
  for (int k = 0; k < D; ++k) {
    gj_factor_kernel<<<1, D, 0, stream>>>(Mre, Mim, fre, fim, k);
    gj_elim_kernel<<<gDD, blk, 0, stream>>>(Mre, Mim, Rre, Rim, fre, fim, k);
  }
  finish_inv_kernel<<<gDD, blk, 0, stream>>>(Mre, Mim, Rre, Rim, Ire, Iim);

  // ---- 3) weight f16 prep ----
  transposeW_f16_kernel<<<gDD, blk, 0, stream>>>(Wre, Wim, WTre, WTim);
  winvcat_f16_kernel<<<(D * D2 + 255) / 256, blk, 0, stream>>>(Ire, Iim, icath);
  cvt_f16_kernel<<<(D2 * D2 + 255) / 256, blk, 0, stream>>>(mix_w, mixh, D2 * D2);
  cvt_f16_kernel<<<(D2 * D2 + 255) / 256, blk, 0, stream>>>(proj_w, projh, D2 * D2);
  tdf_kernel<<<(D + 255) / 256, blk, 0, stream>>>(lam_re, lam_im, tdf);

  const dim3 gGemm(D / 64, MM / 128);   // (8, 128)

  // ---- 4) GEMM1: h = x @ W  (real A, complex B -> 2 planes) ----
  gemm_wmma<1, 2, 0><<<gGemm, blk, 0, stream>>>(
      x, nullptr, WTre, WTim, nullptr, nullptr, h_re, h_im, D, D,
      nullptr, nullptr, nullptr);

  // ---- 5) GEMM2: X = [h_re|h_im] @ mix_w^T + mix_b, split halves ----
  gemm_wmma<2, 2, 0><<<gGemm, blk, 0, stream>>>(
      h_re, h_im, mixh, mixh + (size_t)D * D2, mix_b, mix_b + D,
      X_re, X_im, D2, D, nullptr, nullptr, nullptr);

  // ---- 6) associative scan over T (in place: X -> states) ----
  scan_pass1<<<(BB * D * NCH + 255) / 256, blk, 0, stream>>>(X_re, X_im, dec_re,
                                                             dec_im, carry);
  scan_pass2<<<(BB * D + 255) / 256, blk, 0, stream>>>(dec_re, dec_im, carry);
  scan_pass3<<<(BB * D * NCH + 255) / 256, blk, 0, stream>>>(X_re, X_im, dec_re,
                                                             dec_im, carry);

  // ---- 7) GEMM3: source = [s_re|s_im] @ proj_w^T + proj_b, fused
  //         h_out = h*t_decay + source*forcing (complex epilogue) ----
  gemm_wmma<2, 2, 1><<<gGemm, blk, 0, stream>>>(
      X_re, X_im, projh, projh + (size_t)D * D2, proj_b, proj_b + D,
      ho_re, ho_im, D2, D, h_re, h_im, tdf);

  // ---- 8) GEMM4: x_out = Re(h_out @ Winv) = [ho_re|ho_im] @ [Ire; -Iim] ----
  gemm_wmma<2, 1, 0><<<gGemm, blk, 0, stream>>>(
      ho_re, ho_im, icath, nullptr, nullptr, nullptr, out, nullptr, D2, D,
      nullptr, nullptr, nullptr);
}
